// Recon_block_6201932775519
// MI455X (gfx1250) — matrix operand order repaired and checked by reading, not compiled, not run on hardware
//
#include <hip/hip_runtime.h>

// ---------------- problem constants ----------------
constexpr int BLKSZ  = 48;
constexpr int C      = 31;
constexpr int CP     = 32;          // padded channels
constexpr int NTOK   = BLKSZ * BLKSZ;   // 2304
constexpr int NHB    = 6;           // 288/48 blocks per side
constexpr int NBLOCKS = NHB * NHB;  // 36
constexpr int HIN = 256, WIN = 256;
constexpr int OC1 = 64;

typedef __attribute__((ext_vector_type(16))) _Float16 v16h;
typedef __attribute__((ext_vector_type(8)))  float    v8f;

union V16H { v16h v; uint4 q[2]; _Float16 h[16]; };

// ---------------- 1) reflect-pad + blockify + f16 token-major copy ----------------
__global__ void k_blockify(const float* __restrict__ xt,
                           float* __restrict__ xblk,      // [36][31][2304] f32
                           _Float16* __restrict__ Xh) {   // [36][2304][32] f16, col31=0
  int idx = blockIdx.x * blockDim.x + threadIdx.x;
  if (idx >= NBLOCKS * CP * NTOK) return;
  int t   = idx % NTOK;
  int ch  = (idx / NTOK) % CP;
  int blk = idx / (NTOK * CP);
  int h = t / BLKSZ, w = t % BLKSZ;
  int gh = (blk / NHB) * BLKSZ + h;
  int gw = (blk % NHB) * BLKSZ + w;
  if (gh >= HIN) gh = 2 * HIN - 2 - gh;   // numpy/jax 'reflect'
  if (gw >= WIN) gw = 2 * WIN - 2 - gw;
  if (ch < C) {
    float v = xt[ch * (HIN * WIN) + gh * WIN + gw];
    xblk[((size_t)blk * C + ch) * NTOK + t] = v;
    Xh[((size_t)blk * NTOK + t) * CP + ch] = (_Float16)v;
  } else {
    Xh[((size_t)blk * NTOK + t) * CP + ch] = (_Float16)0.0f;
  }
}

// ---------------- 2) conv1 3x3 (31->64) + relu ----------------
__global__ void k_conv1_relu(const float* __restrict__ xblk,
                             const float* __restrict__ w1,
                             float* __restrict__ y1) {    // [36][64][2304]
  int idx = blockIdx.x * blockDim.x + threadIdx.x;
  if (idx >= NBLOCKS * OC1 * NTOK) return;
  int t   = idx % NTOK;
  int oc  = (idx / NTOK) % OC1;
  int blk = idx / (NTOK * OC1);
  int h = t / BLKSZ, w = t % BLKSZ;
  const float* xb = xblk + (size_t)blk * C * NTOK;
  float acc = 0.f;
  for (int ic = 0; ic < C; ++ic) {
    const float* xc = xb + (size_t)ic * NTOK;
    const float* wk = w1 + ((size_t)oc * C + ic) * 9;
#pragma unroll
    for (int dh = 0; dh < 3; ++dh) {
      int hh = h + dh - 1;
      if (hh < 0 || hh >= BLKSZ) continue;
#pragma unroll
      for (int dw = 0; dw < 3; ++dw) {
        int ww = w + dw - 1;
        if (ww < 0 || ww >= BLKSZ) continue;
        acc += xc[hh * BLKSZ + ww] * wk[dh * 3 + dw];
      }
    }
  }
  y1[idx] = fmaxf(acc, 0.f);
}

// ---------------- 3) conv2 3x3 (64->31) + residual x ----------------
__global__ void k_conv2_add(const float* __restrict__ y1,
                            const float* __restrict__ w2,
                            const float* __restrict__ xblk,
                            float* __restrict__ z1) {     // [36][31][2304]
  int idx = blockIdx.x * blockDim.x + threadIdx.x;
  if (idx >= NBLOCKS * C * NTOK) return;
  int t   = idx % NTOK;
  int oc  = (idx / NTOK) % C;
  int blk = idx / (NTOK * C);
  int h = t / BLKSZ, w = t % BLKSZ;
  const float* yb = y1 + (size_t)blk * OC1 * NTOK;
  float acc = 0.f;
  for (int ic = 0; ic < OC1; ++ic) {
    const float* yc = yb + (size_t)ic * NTOK;
    const float* wk = w2 + ((size_t)oc * OC1 + ic) * 9;
#pragma unroll
    for (int dh = 0; dh < 3; ++dh) {
      int hh = h + dh - 1;
      if (hh < 0 || hh >= BLKSZ) continue;
#pragma unroll
      for (int dw = 0; dw < 3; ++dw) {
        int ww = w + dw - 1;
        if (ww < 0 || ww >= BLKSZ) continue;
        acc += yc[hh * BLKSZ + ww] * wk[dh * 3 + dw];
      }
    }
  }
  z1[idx] = xblk[idx] + acc;
}

// ---------------- 4) conv3 1x1 (31->31) -> V channel-major f16 ----------------
__global__ void k_conv3_v(const float* __restrict__ xblk,
                          const float* __restrict__ w3,
                          _Float16* __restrict__ Vh) {    // [36][32][2304], row31=0
  int idx = blockIdx.x * blockDim.x + threadIdx.x;
  if (idx >= NBLOCKS * CP * NTOK) return;
  int t   = idx % NTOK;
  int oc  = (idx / NTOK) % CP;
  int blk = idx / (NTOK * CP);
  if (oc < C) {
    const float* xb = xblk + (size_t)blk * C * NTOK + t;
    float acc = 0.f;
    for (int ic = 0; ic < C; ++ic)
      acc += w3[oc * C + ic] * xb[(size_t)ic * NTOK];
    Vh[idx] = (_Float16)acc;
  } else {
    Vh[idx] = (_Float16)0.0f;
  }
}

// ---------------- 5) attention: one wave per 16-row strip ----------------
// S strip (16 x 2304 f32) lives in LDS (147KB of CDNA5's 320KB/WGP).
__global__ void __launch_bounds__(32)
k_attn(const _Float16* __restrict__ Xh,    // [36][2304][32]  (Q and K are both x)
       const _Float16* __restrict__ Vh,    // [36][32][2304]
       float* __restrict__ ctx) {          // [36][2304][31]  relu'd & scaled
  extern __shared__ float smem[];
  float* S      = smem;             // 16*2304 floats
  float* rowsum = smem + 16 * NTOK; // 16 floats

  const int wg   = blockIdx.x;
  const int b    = wg / (NTOK / 16);
  const int mb   = (wg % (NTOK / 16)) * 16;
  const int lane = threadIdx.x;
  const int m    = lane & 15;       // A row / B col / D col index
  const int half = lane >> 4;
  const float scaleAttn = 0.17960530202677491f;  // 31^-0.5

  // A fragment: Q rows mb..mb+15 (16x32 f16, ISA interleave: K 0-7,16-23 | 8-15,24-31)
  V16H A;
  {
    const uint4* rq = (const uint4*)(Xh + (size_t)(b * NTOK + mb + m) * CP);
    A.q[0] = rq[half];       // K = half*8 .. half*8+7
    A.q[1] = rq[half + 2];   // K = half*8+16 .. half*8+23
  }

  // ---- S = (Q K^T) * scale, striped into LDS ----
  for (int j = 0; j < NTOK / 16; ++j) {
    const int nb = j * 16;
    V16H B;
    const uint4* rk = (const uint4*)(Xh + (size_t)(b * NTOK + nb + m) * CP);
    B.q[0] = rk[half];       // lane = col m, K = half*8 .. half*8+7
    B.q[1] = rk[half + 2];   // K = half*8+16 .. half*8+23
    v8f c = {};
    c = __builtin_amdgcn_wmma_f32_16x16x32_f16(false, A.v, false, B.v,
                                               (short)0, c, false, false);
#pragma unroll
    for (int r = 0; r < 8; ++r)
      S[(r + half * 8) * NTOK + nb + m] = c[r] * scaleAttn;  // D: M=r+half*8, N=m
  }

  // ---- exact two-pass softmax; lane pair (l, l^16) shares row m ----
  const int cb = half * (NTOK / 2);
  float mx = -3.0e38f;
  for (int i = 0; i < NTOK / 2; ++i)
    mx = fmaxf(mx, S[m * NTOK + cb + i]);
  mx = fmaxf(mx, __shfl_xor(mx, 16, 32));
  float sum = 0.f;
  for (int i = 0; i < NTOK / 2; ++i) {
    float e = __expf(S[m * NTOK + cb + i] - mx);
    S[m * NTOK + cb + i] = e;        // keep unnormalized P; normalize at the end
    sum += e;
  }
  sum += __shfl_xor(sum, 16, 32);
  if (lane < 16) rowsum[m] = sum;

  // ---- ctx = P V  (K chunks of 32 columns; N split 0-15 / 16-31) ----
  v8f acc0 = {}, acc1 = {};
  for (int jj = 0; jj < NTOK / 32; ++jj) {
    V16H P;
#pragma unroll
    for (int v = 0; v < 8; ++v) {
      int k0 = half * 16 + 2 * v;
      float2 f = *(const float2*)&S[m * NTOK + jj * 32 + k0];
      P.h[2 * v]     = (_Float16)f.x;
      P.h[2 * v + 1] = (_Float16)f.y;
    }
    V16H B0, B1;
    const uint4* rv0 = (const uint4*)(Vh + ((size_t)b * CP + m)      * NTOK + jj * 32 + half * 16);
    const uint4* rv1 = (const uint4*)(Vh + ((size_t)b * CP + 16 + m) * NTOK + jj * 32 + half * 16);
    B0.q[0] = rv0[0]; B0.q[1] = rv0[1];
    B1.q[0] = rv1[0]; B1.q[1] = rv1[1];
    acc0 = __builtin_amdgcn_wmma_f32_16x16x32_f16(false, P.v, false, B0.v,
                                                  (short)0, acc0, false, false);
    acc1 = __builtin_amdgcn_wmma_f32_16x16x32_f16(false, P.v, false, B1.v,
                                                  (short)0, acc1, false, false);
  }

  // ---- normalize, * 48/78, relu, store (natural (N, c) buffer) ----
  const float ctxScale = (float)BLKSZ / (float)(BLKSZ + C - 1);
  float* cbout = ctx + (size_t)b * NTOK * C;
#pragma unroll
  for (int r = 0; r < 8; ++r) {
    int mrow = r + half * 8;
    float inv = ctxScale / rowsum[mrow];
    size_t grow = (size_t)(mb + mrow) * C;
    cbout[grow + m] = fmaxf(acc0[r] * inv, 0.f);
    if (m < 15)  // col 16+m; col 31 doesn't exist
      cbout[grow + 16 + m] = fmaxf(acc1[r] * inv, 0.f);
  }
}

// ---------------- 6) blend (raw (N,c)->(c,48,48) reinterpret), unblock, crop ----------------
__global__ void k_epilogue(const float* __restrict__ z1,
                           const float* __restrict__ ctx,
                           const float* __restrict__ wz1,
                           float* __restrict__ out) {
  int idx = blockIdx.x * blockDim.x + threadIdx.x;
  if (idx >= C * HIN * WIN) return;
  int ch  = idx / (HIN * WIN);
  int rem = idx % (HIN * WIN);
  int gh = rem / WIN, gw = rem % WIN;
  int blk = (gh / BLKSZ) * NHB + (gw / BLKSZ);
  int h = gh % BLKSZ, w = gw % BLKSZ;
  int f  = ch * NTOK + h * BLKSZ + w;   // flat index into (c,48,48) view
  int n  = f / C, c2 = f - n * C;       // raw reinterpretation of (N, c) buffer
  float cv = ctx[((size_t)blk * NTOK + n) * C + c2];          // already relu'd+scaled
  float zv = z1[((size_t)blk * C + ch) * NTOK + h * BLKSZ + w];
  float wz = wz1[0];
  out[idx] = wz * zv + (1.f - wz) * cv;
}

// ---------------- host launcher ----------------
extern "C" void kernel_launch(void* const* d_in, const int* in_sizes, int n_in,
                              void* d_out, int out_size, void* d_ws, size_t ws_size,
                              hipStream_t stream) {
  const float* xt  = (const float*)d_in[0];
  const float* w1  = (const float*)d_in[1];
  const float* w2  = (const float*)d_in[2];
  const float* w3  = (const float*)d_in[3];
  const float* wz1 = (const float*)d_in[4];

  char* p = (char*)d_ws;
  auto alloc = [&](size_t bytes) {
    char* r = p; p += (bytes + 255) & ~(size_t)255; return (void*)r;
  };
  float*    xblk = (float*)   alloc((size_t)NBLOCKS * C   * NTOK * 4);
  _Float16* Xh   = (_Float16*)alloc((size_t)NBLOCKS * NTOK * CP  * 2);
  float*    y1   = (float*)   alloc((size_t)NBLOCKS * OC1 * NTOK * 4);
  float*    z1   = (float*)   alloc((size_t)NBLOCKS * C   * NTOK * 4);
  _Float16* Vh   = (_Float16*)alloc((size_t)NBLOCKS * CP  * NTOK * 2);
  float*    ctx  = (float*)   alloc((size_t)NBLOCKS * NTOK * C   * 4);

  k_blockify  <<<(NBLOCKS * CP  * NTOK + 255) / 256, 256, 0, stream>>>(xt, xblk, Xh);
  k_conv1_relu<<<(NBLOCKS * OC1 * NTOK + 255) / 256, 256, 0, stream>>>(xblk, w1, y1);
  k_conv2_add <<<(NBLOCKS * C   * NTOK + 255) / 256, 256, 0, stream>>>(y1, w2, xblk, z1);
  k_conv3_v   <<<(NBLOCKS * CP  * NTOK + 255) / 256, 256, 0, stream>>>(xblk, w3, Vh);

  size_t lds = (size_t)(16 * NTOK + 16) * sizeof(float);  // 147,520 B < 320KB/WGP
  k_attn<<<NBLOCKS * (NTOK / 16), 32, lds, stream>>>(Xh, Vh, ctx);

  k_epilogue<<<(C * HIN * WIN + 255) / 256, 256, 0, stream>>>(z1, ctx, wz1, (float*)d_out);
}
